// MultiHeadAttention_9088150798723
// MI455X (gfx1250) — matrix operand order repaired and checked by reading, not compiled, not run on hardware
//
#include <hip/hip_runtime.h>

#define S_LEN 4096
#define DMODEL 1024
#define NHEAD 16
#define DK 64
#define NEGBIG (-1.0e9f)

typedef __attribute__((ext_vector_type(16))) _Float16 v16h;
typedef __attribute__((ext_vector_type(8)))  _Float16 v8h;
typedef __attribute__((ext_vector_type(8)))  float    v8f;
typedef __attribute__((ext_vector_type(4)))  float    v4f;

union F16x16 { v16h v; v8h h[2]; };

static __device__ __forceinline__ int iclamp(int x, int lo, int hi) {
  return x < lo ? lo : (x > hi ? hi : x);
}

static __device__ __forceinline__ v8f wmma16(v16h a, v16h b, v8f c) {
  // D = A(16x32 f16) * B(32x16 f16) + C(16x16 f32)
  return __builtin_amdgcn_wmma_f32_16x16x32_f16(false, a, false, b, (short)0, c,
                                                false, false);
}

// A fragment 16x32 (MxK) from row-major f16 matrix X with leading dim ld.
// Lane l holds row m0+(l&15); halves: K = k0+8*(l>>4)+{0..7} and +16.
static __device__ __forceinline__ v16h load_a(const _Float16* X, int ld, int m0,
                                              int k0, int lane) {
  const _Float16* p =
      X + (size_t)(m0 + (lane & 15)) * (size_t)ld + (size_t)(k0 + ((lane >> 4) << 3));
  F16x16 f;
  f.h[0] = *(const v8h*)p;
  f.h[1] = *(const v8h*)(p + 16);
  return f.v;
}

// B fragment 32x16 (KxN) where B(k,n) = R[n0+n][k0+k], R row-major, ld leading.
// Lane l holds col n0+(l&15); halves: K = k0+8*(l>>4)+{0..7} and +16.
static __device__ __forceinline__ v16h load_b(const _Float16* R, int ld, int n0,
                                              int k0, int lane) {
  const _Float16* p =
      R + (size_t)(n0 + (lane & 15)) * (size_t)ld + (size_t)(k0 + ((lane >> 4) << 3));
  F16x16 f;
  f.h[0] = *(const v8h*)p;
  f.h[1] = *(const v8h*)(p + 16);
  return f.v;
}

// 8 elements per thread: 2x b128 loads, convert, 1x b128 store.
__global__ void cvt_f32_f16(const float* __restrict__ in, _Float16* __restrict__ out,
                            int n8) {
  int i = blockIdx.x * blockDim.x + threadIdx.x;
  if (i < n8) {
    v4f a = ((const v4f*)in)[2 * i];
    v4f b = ((const v4f*)in)[2 * i + 1];
    v8h o;
#pragma unroll
    for (int r = 0; r < 4; ++r) o[r] = (_Float16)a[r];
#pragma unroll
    for (int r = 0; r < 4; ++r) o[4 + r] = (_Float16)b[r];
    ((v8h*)out)[i] = o;
  }
}

// Y[m, n] = sum_k X[m,k] * W[n,k] (+ bias[n]).  M=S_LEN, N=K=DMODEL.
// One wave computes a 32x64 output tile: 8 WMMAs per K-step on 4 shared B frags.
// MODE 0: f16 out[m*N + n]   (row-major, for Q/K)
// MODE 1: f16 out[n*M + m]   (transposed, for V -> Vt[d_global][s]; vector store)
// MODE 2: f32 out[m*N + n]   (final output projection)
template <int MODE>
__global__ __launch_bounds__(128) void proj_gemm(const _Float16* __restrict__ X,
                                                 const _Float16* __restrict__ W,
                                                 const float* __restrict__ bias,
                                                 void* __restrict__ out) {
  const int lane = threadIdx.x & 31;
  const int wave = threadIdx.x >> 5;                 // 0..3
  const int m0 = blockIdx.y * 32;
  const int n0 = blockIdx.x * 256 + wave * 64;       // 4 N-tiles of 16 per wave

  v8f acc[2][4];
#pragma unroll
  for (int t = 0; t < 2; ++t)
#pragma unroll
    for (int j = 0; j < 4; ++j) acc[t][j] = v8f{};

  for (int k0 = 0; k0 < DMODEL; k0 += 32) {
    v16h a0 = load_a(X, DMODEL, m0, k0, lane);
    v16h a1 = load_a(X, DMODEL, m0 + 16, k0, lane);
#pragma unroll
    for (int j = 0; j < 4; ++j) {
      v16h b = load_b(W, DMODEL, n0 + 16 * j, k0, lane);
      if (k0 + 32 < DMODEL) {
        // pull next K-step's weight fragment toward the WGP (global_prefetch_b8)
        __builtin_prefetch(W + (size_t)(n0 + 16 * j + (lane & 15)) * DMODEL +
                               (k0 + 32 + ((lane >> 4) << 4)),
                           0, 3);
      }
      acc[0][j] = wmma16(a0, b, acc[0][j]);
      acc[1][j] = wmma16(a1, b, acc[1][j]);
    }
  }

  const int nl = lane & 15;
  const int hi8 = 8 * (lane >> 4);
#pragma unroll
  for (int t = 0; t < 2; ++t) {
    const int mbase = m0 + 16 * t + hi8;
#pragma unroll
    for (int j = 0; j < 4; ++j) {
      int n = n0 + 16 * j + nl;
      float bv = bias[n];
      if (MODE == 1) {
        // Vt[n][mbase..mbase+7] is contiguous -> single 16B store
        v8h pk;
#pragma unroll
        for (int r = 0; r < 8; ++r) pk[r] = (_Float16)(acc[t][j][r] + bv);
        *(v8h*)&((_Float16*)out)[(size_t)n * S_LEN + mbase] = pk;
      } else {
#pragma unroll
        for (int r = 0; r < 8; ++r) {
          float val = acc[t][j][r] + bv;
          if (MODE == 0) {
            ((_Float16*)out)[(size_t)(mbase + r) * DMODEL + n] = (_Float16)val;
          } else {
            ((float*)out)[(size_t)(mbase + r) * DMODEL + n] = val;
          }
        }
      }
    }
  }
}

// One wave per (head, 16-row query tile). Flash-style windowed attention.
// Q,K: f16 [S_LEN, DMODEL] (d_global = h*64+d).  Vt: f16 [DMODEL, S_LEN].
// ctx: f16 [S_LEN, DMODEL].
__global__ __launch_bounds__(128) void attn_kernel(const _Float16* __restrict__ Q,
                                                   const _Float16* __restrict__ K,
                                                   const _Float16* __restrict__ Vt,
                                                   _Float16* __restrict__ ctx) {
  __shared__ __align__(16) _Float16 plds[4][16 * 32];

  const int lane = threadIdx.x & 31;
  const int wave = threadIdx.x >> 5;
  const int tile = blockIdx.x * 4 + wave;   // 0..4095
  const int h = tile >> 8;                  // 256 q-tiles per head
  const int q0 = (tile & 255) * 16;
  const int d0 = h * DK;
  const int nl = lane & 15;
  const int hi = lane >> 4;
  _Float16* myl = plds[wave];

  // Q A-fragments for the whole tile (d = 0..31 and 32..63 of this head)
  v16h qa0 = load_a(Q, DMODEL, q0, d0, lane);
  v16h qa1 = load_a(Q, DMODEL, q0, d0 + 32, lane);

  v8f o[4] = {v8f{}, v8f{}, v8f{}, v8f{}};
  float mrow[8], lrow[8];
#pragma unroll
  for (int r = 0; r < 8; ++r) { mrow[r] = -1e30f; lrow[r] = 0.0f; }

  const float scale = 0.125f;  // 1/sqrt(64)

  for (int p = 0; p < 9; ++p) {
    const int k0a = q0 - 128 + 32 * p;  // pair of key tiles: k0a, k0a+16
    const int kr0 = iclamp(k0a, 0, S_LEN - 16);
    const int kr1 = iclamp(k0a + 16, 0, S_LEN - 16);

    // scores = Q . K^T for two 16-key tiles (d split 32+32)
    v8f s0 = v8f{}, s1 = v8f{};
    {
      v16h kb;
      kb = load_b(K, DMODEL, kr0, d0, lane);       s0 = wmma16(qa0, kb, s0);
      kb = load_b(K, DMODEL, kr0, d0 + 32, lane);  s0 = wmma16(qa1, kb, s0);
      kb = load_b(K, DMODEL, kr1, d0, lane);       s1 = wmma16(qa0, kb, s1);
      kb = load_b(K, DMODEL, kr1, d0 + 32, lane);  s1 = wmma16(qa1, kb, s1);
    }

    // scale + window mask (+ sequence bounds for the clamped edge tiles)
#pragma unroll
    for (int r = 0; r < 8; ++r) {
      int sq = q0 + r + 8 * hi;
      int sk0 = k0a + nl;
      int sk1 = sk0 + 16;
      bool ok0 = (sk0 >= sq - 127) && (sk0 <= sq + 128) && (sk0 >= 0) && (sk0 < S_LEN);
      bool ok1 = (sk1 >= sq - 127) && (sk1 <= sq + 128) && (sk1 >= 0) && (sk1 < S_LEN);
      s0[r] = ok0 ? s0[r] * scale : NEGBIG;
      s1[r] = ok1 ? s1[r] * scale : NEGBIG;
    }

    // online softmax per row; rows live in 16-lane halves -> xor-shuffle reduce
#pragma unroll
    for (int r = 0; r < 8; ++r) {
      float t = fmaxf(s0[r], s1[r]);
      t = fmaxf(t, __shfl_xor(t, 1));
      t = fmaxf(t, __shfl_xor(t, 2));
      t = fmaxf(t, __shfl_xor(t, 4));
      t = fmaxf(t, __shfl_xor(t, 8));
      float mn = fmaxf(mrow[r], t);
      float f = __expf(mrow[r] - mn);
      float p0 = __expf(s0[r] - mn);
      float p1 = __expf(s1[r] - mn);
      float ls = p0 + p1;
      ls += __shfl_xor(ls, 1);
      ls += __shfl_xor(ls, 2);
      ls += __shfl_xor(ls, 4);
      ls += __shfl_xor(ls, 8);
      lrow[r] = lrow[r] * f + ls;
      mrow[r] = mn;
      s0[r] = p0;
      s1[r] = p1;
#pragma unroll
      for (int j = 0; j < 4; ++j) o[j][r] *= f;
    }

    // C-layout -> LDS 16x32 probability tile (f16)
#pragma unroll
    for (int r = 0; r < 8; ++r) {
      int m = r + 8 * hi;
      myl[m * 32 + nl]      = (_Float16)s0[r];
      myl[m * 32 + 16 + nl] = (_Float16)s1[r];
    }
    asm volatile("s_wait_dscnt 0" ::: "memory");

    // reload as A-fragment (16 rows x 32 keys)
    F16x16 pa;
    pa.h[0] = *(const v8h*)&myl[(lane & 15) * 32 + hi * 8];
    pa.h[1] = *(const v8h*)&myl[(lane & 15) * 32 + 16 + hi * 8];

    // out += P . V  (Vt rows are d_global, contiguous in s)
    const int vs = iclamp(k0a, 0, S_LEN - 32);  // clamped keys have p == 0 exactly
#pragma unroll
    for (int j = 0; j < 4; ++j) {
      v16h vb = load_b(Vt, S_LEN, d0 + 16 * j, vs, lane);
      o[j] = wmma16(pa.v, vb, o[j]);
    }
  }

  // normalize and store context [s, h*64+d]
#pragma unroll
  for (int r = 0; r < 8; ++r) {
    float inv = 1.0f / lrow[r];
    int m = q0 + r + 8 * hi;
#pragma unroll
    for (int j = 0; j < 4; ++j) {
      ctx[(size_t)m * DMODEL + d0 + 16 * j + nl] = (_Float16)(o[j][r] * inv);
    }
  }
}

extern "C" void kernel_launch(void* const* d_in, const int* in_sizes, int n_in,
                              void* d_out, int out_size, void* d_ws, size_t ws_size,
                              hipStream_t stream) {
  (void)in_sizes; (void)n_in; (void)out_size; (void)ws_size;
  const float* q   = (const float*)d_in[0];
  const float* k   = (const float*)d_in[1];
  const float* v   = (const float*)d_in[2];
  const float* w_q = (const float*)d_in[3];
  const float* b_q = (const float*)d_in[4];
  const float* w_k = (const float*)d_in[5];
  const float* b_k = (const float*)d_in[6];
  const float* w_v = (const float*)d_in[7];
  const float* b_v = (const float*)d_in[8];
  const float* w_o = (const float*)d_in[9];
  const float* b_o = (const float*)d_in[10];

  const size_t NQ = (size_t)S_LEN * DMODEL;   // 4,194,304
  const size_t NW = (size_t)DMODEL * DMODEL;  // 1,048,576

  _Float16* base = (_Float16*)d_ws;
  _Float16* qh  = base;                 // f16 copies of inputs
  _Float16* kh  = base + NQ;
  _Float16* vh  = base + 2 * NQ;
  _Float16* wqh = base + 3 * NQ;
  _Float16* wkh = base + 3 * NQ + NW;
  _Float16* wvh = base + 3 * NQ + 2 * NW;
  _Float16* woh = base + 3 * NQ + 3 * NW;
  _Float16* Qb  = base + 3 * NQ + 4 * NW;  // projected Q  [S, DMODEL]
  _Float16* Kb  = base + 4 * NQ + 4 * NW;  // projected K  [S, DMODEL]
  _Float16* Vtb = base + 5 * NQ + 4 * NW;  // projected V^T [DMODEL, S]
  _Float16* ctx = base + 6 * NQ + 4 * NW;  // attention output [S, DMODEL]
  // total workspace use: 7*NQ + 4*NW halves = 64 MB

  auto cvt = [&](const float* src, _Float16* dst, size_t n) {
    int n8 = (int)(n / 8);
    cvt_f32_f16<<<dim3((unsigned)((n8 + 255) / 256)), dim3(256), 0, stream>>>(
        src, dst, n8);
  };
  cvt(q, qh, NQ);
  cvt(k, kh, NQ);
  cvt(v, vh, NQ);
  cvt(w_q, wqh, NW);
  cvt(w_k, wkh, NW);
  cvt(w_v, wvh, NW);
  cvt(w_o, woh, NW);

  dim3 pgrid(DMODEL / 256, S_LEN / 32);
  proj_gemm<0><<<pgrid, 128, 0, stream>>>(qh, wqh, b_q, (void*)Qb);
  proj_gemm<0><<<pgrid, 128, 0, stream>>>(kh, wkh, b_k, (void*)Kb);
  proj_gemm<1><<<pgrid, 128, 0, stream>>>(vh, wvh, b_v, (void*)Vtb);

  attn_kernel<<<dim3((NHEAD * (S_LEN / 16)) / 4), 128, 0, stream>>>(Qb, Kb, Vtb, ctx);

  proj_gemm<2><<<pgrid, 128, 0, stream>>>(ctx, woh, b_o, d_out);
}
